// GCN_17592186044980
// MI455X (gfx1250) — matrix operand order repaired and checked by reading, not compiled, not run on hardware
//
#include <hip/hip_runtime.h>
#include <hip/hip_bf16.h>

#define D 128

typedef __attribute__((ext_vector_type(16))) __bf16 v16bf;
typedef __attribute__((ext_vector_type(8)))  float  v8f;
typedef __attribute__((ext_vector_type(4)))  unsigned int u32x4;
typedef __attribute__((ext_vector_type(4)))  int  i32x4;
typedef __attribute__((ext_vector_type(8)))  int  i32x8;

// round-to-nearest-even f32 -> bf16 bit pattern
__device__ __forceinline__ unsigned short f2bf(float f) {
    union { float f; unsigned int u; } v; v.f = f;
    unsigned int u = v.u;
    unsigned int r = u + 0x7FFFu + ((u >> 16) & 1u);
    return (unsigned short)(r >> 16);
}

// ---------------------------------------------------------------------------
// Tensor Data Mover: 1-D DMA of `bytes` (multiple of 8) from global -> LDS.
// D# group0: count=1 | lds_addr | global_addr[56:0] | type=2 ("image").
// D# group1: data_size=3 (8B elems), tensor_dim0 = tile_dim0 = bytes/8,
//            tensor_dim1 = 1, groups 2/3 zero (<=2D tensor).
// ---------------------------------------------------------------------------
__device__ __forceinline__ void tdm_load_to_lds(unsigned lds_off, const void* g, unsigned bytes) {
    unsigned long long ga = (unsigned long long)(size_t)g;
    unsigned n8 = bytes >> 3;
    u32x4 g0;
    g0[0] = 1u;                                             // count=1, user mode
    g0[1] = lds_off;                                        // LDS byte address
    g0[2] = (unsigned)ga;                                   // global addr lo
    g0[3] = ((unsigned)(ga >> 32) & 0x01FFFFFFu) | (2u << 30); // addr hi | type=2
    i32x8 g1;
    g1[0] = 3 << 16;                                        // data_size = 8B
    g1[1] = (int)((n8 & 0xFFFFu) << 16);                    // tensor_dim0[15:0]
    g1[2] = (int)(((n8 >> 16) & 0xFFFFu) | (1u << 16));     // tensor_dim0 hi | tensor_dim1=1
    g1[3] = (int)((n8 & 0xFFFFu) << 16);                    // tile_dim0
    g1[4] = 0;                                              // tile_dim1/2 unused
    g1[5] = (int)n8;                                        // tensor_dim0_stride lo
    g1[6] = 0;
    g1[7] = 0;
    i32x4 zz = {0, 0, 0, 0};
#if defined(__clang_major__) && (__clang_major__ >= 23)
    i32x8 z8 = {0, 0, 0, 0, 0, 0, 0, 0};
    __builtin_amdgcn_tensor_load_to_lds(g0, g1, zz, zz, z8, 0);
#else
    __builtin_amdgcn_tensor_load_to_lds(g0, g1, zz, zz, 0);
#endif
}

// ---------------------------------------------------------------------------
// One-shot weight prep: f32 [128][128] row-major -> bf16 in WMMA B-fragment
// order [c(8)][kt(4)][lane(32)][e(16)] so GEMM blocks DMA it straight to LDS.
// ---------------------------------------------------------------------------
__global__ __launch_bounds__(256) void prep_weight(const float* __restrict__ W,
                                                   unsigned short* __restrict__ out)
{
    int f = blockIdx.x * 256 + threadIdx.x;   // 0..16383
    int k = f >> 7;
    int n = f & 127;
    float v = W[k * D + n];
    int c  = n >> 4, nl = n & 15;
    int kt = k >> 5, kk = k & 31;
    int e  = (kk & 7) + ((kk >> 4) << 3);
    int l  = nl + 16 * ((kk >> 3) & 1);
    out[((c * 4 + kt) * 32 + l) * 16 + e] = f2bf(v);
}

// ---------------------------------------------------------------------------
// Fused up-to-3-weight GEMM: O_w = X @ W_w (+bias0 if given).
// Block: 256 threads (8 waves). Tile: 32 rows x 128 cols per block.
// Weights arrive pre-swizzled bf16; staged to LDS via TDM (4KB per wave).
// ---------------------------------------------------------------------------
__global__ __launch_bounds__(256) void gemm3_wmma(
    const float* __restrict__ X,
    const unsigned short* __restrict__ Wb0, const unsigned short* __restrict__ Wb1,
    const unsigned short* __restrict__ Wb2,
    float* __restrict__ O0, float* __restrict__ O1, float* __restrict__ O2,
    const float* __restrict__ bias0, int nrows, int nw)
{
    __shared__ __align__(32) unsigned short lA[2 * 4 * 32 * 16];   //  8 KB
    __shared__ __align__(32) unsigned short lB[8 * 4 * 32 * 16];   // 32 KB

    const int tid     = threadIdx.x;
    const int lane    = tid & 31;
    const int wave    = tid >> 5;
    const int rt      = wave >> 2;      // row-tile of this wave (0/1)
    const int cg      = wave & 3;       // column group (32 cols each)
    const int rowbase = blockIdx.x * 32;

    if (nw > 1 && tid == 0) {
        __builtin_prefetch(Wb1, 0, 0);
        __builtin_prefetch(Wb2, 0, 0);
    }

    // ---- load + swizzle A tile (32 rows x 128 K) into fragment order ----
    #pragma unroll
    for (int i = 0; i < 16; ++i) {
        int f   = i * 256 + tid;          // 0..4095
        int r   = f >> 7;                 // 0..31
        int k   = f & 127;
        int row = rowbase + r;
        float v = (row < nrows) ? X[row * D + k] : 0.0f;
        int art = r >> 4, m = r & 15;
        int kt  = k >> 5, kk = k & 31;
        int lhalf = (kk >> 3) & 1;
        int e     = (kk & 7) + (((kk >> 4) & 1) << 3);
        int l     = m + 16 * lhalf;
        lA[(((art * 4) + kt) * 32 + l) * 16 + e] = f2bf(v);
    }
    __syncthreads();

    // ---- this wave's 4 A fragments (K = 0..127 in steps of 32) ----
    v16bf af[4];
    #pragma unroll
    for (int kt = 0; kt < 4; ++kt)
        af[kt] = *reinterpret_cast<const v16bf*>(&lA[((rt * 4 + kt) * 32 + lane) * 16]);

    const unsigned ldsB = (unsigned)(size_t)(void*)lB;
    const unsigned wofs = (unsigned)__builtin_amdgcn_readfirstlane((int)(wave << 12)); // wave*4096B

    for (int w = 0; w < nw; ++w) {
        const unsigned short* Wbp = (w == 0) ? Wb0 : ((w == 1) ? Wb1 : Wb2);
        float*                Op  = (w == 0) ? O0  : ((w == 1) ? O1  : O2);

        __syncthreads();   // previous lB fully consumed
        // ---- async-tensor DMA: each wave moves a 4KB slice of W to LDS ----
        tdm_load_to_lds(ldsB + wofs, (const void*)(Wbp + (wofs >> 1)), 4096);
        __builtin_amdgcn_s_wait_tensorcnt(0);
        __syncthreads();

        #pragma unroll
        for (int ct = 0; ct < 2; ++ct) {
            int c = cg * 2 + ct;          // 16-col tile index (0..7)
            v8f acc = {};
            #pragma unroll
            for (int kt = 0; kt < 4; ++kt) {
                v16bf bf = *reinterpret_cast<const v16bf*>(&lB[((c * 4 + kt) * 32 + lane) * 16]);
                acc = __builtin_amdgcn_wmma_f32_16x16x32_bf16(
                    false, af[kt], false, bf, (short)0, acc, false, false);
            }
            // ---- store D tile (f32 16x16 C/D layout) ----
            int coln  = c * 16 + (lane & 15);
            int rbase = rowbase + rt * 16 + ((lane >> 4) & 1) * 8;
            float badd = (bias0 != nullptr) ? bias0[coln] : 0.0f;
            if (rbase + 8 <= nrows) {
                #pragma unroll
                for (int r = 0; r < 8; ++r)
                    Op[(rbase + r) * D + coln] = acc[r] + badd;
            } else {
                #pragma unroll
                for (int r = 0; r < 8; ++r) {
                    int row = rbase + r;
                    if (row < nrows) Op[row * D + coln] = acc[r] + badd;
                }
            }
        }
    }
}

// ---------------------------------------------------------------------------
// Edge scatter-add: one wave per edge, one float4 per lane, f32 HW atomics.
// ---------------------------------------------------------------------------
__global__ void scatter_add(const float* __restrict__ m, const int* __restrict__ src,
                            const int* __restrict__ dst, float* __restrict__ agg,
                            int nedges)
{
    long long total  = (long long)nedges * 32;
    long long stride = (long long)gridDim.x * blockDim.x;
    for (long long t = (long long)blockIdx.x * blockDim.x + threadIdx.x; t < total; t += stride) {
        int e = (int)(t >> 5);
        int l = (int)(t & 31);
        int s = src[e];
        int d = dst[e];
        float4 v = reinterpret_cast<const float4*>(m + (size_t)s * D)[l];
        float* p = agg + (size_t)d * D + l * 4;
        unsafeAtomicAdd(p + 0, v.x);
        unsafeAtomicAdd(p + 1, v.y);
        unsafeAtomicAdd(p + 2, v.z);
        unsafeAtomicAdd(p + 3, v.w);
    }
}

__global__ void degree_accum(const int* __restrict__ dst, float* __restrict__ deg, int nedges)
{
    long long stride = (long long)gridDim.x * blockDim.x;
    for (long long t = (long long)blockIdx.x * blockDim.x + threadIdx.x; t < nedges; t += stride)
        unsafeAtomicAdd(&deg[dst[t]], 1.0f);
}

__global__ void inv_clamp(float* __restrict__ p, int n)
{
    long long stride = (long long)gridDim.x * blockDim.x;
    for (long long t = (long long)blockIdx.x * blockDim.x + threadIdx.x; t < n; t += stride)
        p[t] = 1.0f / fmaxf(p[t], 1.0f);
}

__global__ void fill_zero(float* __restrict__ p, long long n)
{
    long long stride = (long long)gridDim.x * blockDim.x;
    for (long long t = (long long)blockIdx.x * blockDim.x + threadIdx.x; t < n; t += stride)
        p[t] = 0.0f;
}

// out = relu(agga*inva[row] + ba[col] + aggb*invb[row] + bb[col] + tl)
__global__ void finalize_k(const float* __restrict__ agga, const float* __restrict__ aggb,
                           const float* __restrict__ tl,
                           const float* __restrict__ inva, const float* __restrict__ invb,
                           const float* __restrict__ ba, const float* __restrict__ bb,
                           float* __restrict__ out, int n)
{
    long long total  = (long long)n * D;
    long long stride = (long long)gridDim.x * blockDim.x;
    for (long long t = (long long)blockIdx.x * blockDim.x + threadIdx.x; t < total; t += stride) {
        int i = (int)(t >> 7);
        int j = (int)(t & 127);
        float v = agga[t] * inva[i] + ba[j] + aggb[t] * invb[i] + bb[j] + tl[t];
        out[t] = fmaxf(v, 0.0f);
    }
}

extern "C" void kernel_launch(void* const* d_in, const int* in_sizes, int n_in,
                              void* d_out, int out_size, void* d_ws, size_t ws_size,
                              hipStream_t stream)
{
    const float* x      = (const float*)d_in[0];
    const float* W_proj = (const float*)d_in[1];
    const float* b_proj = (const float*)d_in[2];
    const float* W1a    = (const float*)d_in[3];
    const float* b1a    = (const float*)d_in[4];
    const float* W1b    = (const float*)d_in[5];
    const float* b1b    = (const float*)d_in[6];
    const float* loop1  = (const float*)d_in[7];
    const float* W2a    = (const float*)d_in[8];
    const float* b2a    = (const float*)d_in[9];
    const float* W2b    = (const float*)d_in[10];
    const float* b2b    = (const float*)d_in[11];
    const float* loop2  = (const float*)d_in[12];
    const int*   srca   = (const int*)d_in[13];
    const int*   dsta   = (const int*)d_in[14];
    const int*   srcb   = (const int*)d_in[15];
    const int*   dstb   = (const int*)d_in[16];

    const int N  = in_sizes[0] / D;
    const int Ea = in_sizes[13];
    const int Eb = in_sizes[15];

    float* out = (float*)d_out;
    float* ws  = (float*)d_ws;
    const size_t ND = (size_t)N * D;

    float* h    = ws;
    float* ta   = ws + ND;
    float* tb   = ws + 2 * ND;
    float* agga = ws + 3 * ND;
    float* aggb = ws + 4 * ND;
    float* inva = ws + 5 * ND;
    float* invb = inva + N;
    unsigned short* wbf = reinterpret_cast<unsigned short*>(invb + N);  // 7 x 16384 bf16

    const int T = 256;
    auto blocks = [](long long n, int t) { return (int)((n + t - 1) / t); };
    const int gGemm = (N + 31) / 32;
    const int gScA  = blocks((long long)Ea * 32, T);
    const int gScB  = blocks((long long)Eb * 32, T);
    const int gND   = blocks((long long)ND, T);
    const int WSZ   = 16384;   // shorts per prepped weight

    // ---- one-shot weight convert+swizzle into fragment order ----
    const float* wsrc[7] = {W_proj, W1a, W1b, loop1, W2a, W2b, loop2};
    for (int i = 0; i < 7; ++i)
        prep_weight<<<WSZ / T, T, 0, stream>>>(wsrc[i], wbf + (size_t)i * WSZ);

    // ---- degrees ----
    fill_zero<<<blocks(2 * (long long)N, T), T, 0, stream>>>(inva, 2 * (long long)N);
    degree_accum<<<blocks(Ea, T), T, 0, stream>>>(dsta, inva, Ea);
    degree_accum<<<blocks(Eb, T), T, 0, stream>>>(dstb, invb, Eb);
    inv_clamp<<<blocks(N, T), T, 0, stream>>>(inva, N);
    inv_clamp<<<blocks(N, T), T, 0, stream>>>(invb, N);

    // ---- projection: h = x @ W_proj + b_proj ----
    gemm3_wmma<<<gGemm, T, 0, stream>>>(x, wbf + 0 * WSZ, nullptr, nullptr,
                                        h, nullptr, nullptr, b_proj, N, 1);

    // ---- layer 1 ----
    gemm3_wmma<<<gGemm, T, 0, stream>>>(h, wbf + 1 * WSZ, wbf + 2 * WSZ, wbf + 3 * WSZ,
                                        ta, tb, out /*self-loop scratch*/, nullptr, N, 3);
    fill_zero<<<gND, T, 0, stream>>>(agga, (long long)ND);
    fill_zero<<<gND, T, 0, stream>>>(aggb, (long long)ND);
    scatter_add<<<gScA, T, 0, stream>>>(ta, srca, dsta, agga, Ea);
    scatter_add<<<gScB, T, 0, stream>>>(tb, srcb, dstb, aggb, Eb);
    finalize_k<<<gND, T, 0, stream>>>(agga, aggb, out, inva, invb, b1a, b1b, h, N);

    // ---- layer 2 ----
    gemm3_wmma<<<gGemm, T, 0, stream>>>(h, wbf + 4 * WSZ, wbf + 5 * WSZ, wbf + 6 * WSZ,
                                        ta, tb, out, nullptr, N, 3);
    fill_zero<<<gND, T, 0, stream>>>(agga, (long long)ND);
    fill_zero<<<gND, T, 0, stream>>>(aggb, (long long)ND);
    scatter_add<<<gScA, T, 0, stream>>>(ta, srca, dsta, agga, Ea);
    scatter_add<<<gScB, T, 0, stream>>>(tb, srcb, dstb, aggb, Eb);
    finalize_k<<<gND, T, 0, stream>>>(agga, aggb, out, inva, invb, b2a, b2b, out, N);
}
